// ScaledDotProductAttention_14242111554175
// MI455X (gfx1250) — matrix operand order repaired and checked by reading, not compiled, not run on hardware
//
#include <hip/hip_runtime.h>
#include <hip/hip_bf16.h>

// ---------------------------------------------------------------------------
// Causal flash-attention forward for B=4, H=16, S=2048, D=128 (fp32 in/out).
// Math runs on v_wmma_f32_16x16x32_f16 (CDNA5 / gfx1250, wave32).
//
// Tiling:
//   grid = (S/128, B*H), block = 256 threads = 8 waves.
//   Each wave owns 16 query rows; the block shares 64-key K/V tiles in LDS.
//
// Matmul orientation (avoids any P transpose):
//   S^T = K * Q^T      (A = K tile 16x32-chunk, B = Q^T chunk, C: M=key, N=query)
//   O^T = V^T * P^T    (A = V^T tile,           B = P^T,       C: M=d,   N=query)
// With the C layout (lane = column = query), softmax is per-lane + one
// shfl_xor(16); P converts to the next B-fragment with shfl_xor(16) + selects.
// Causal masking only runs on diagonal key blocks (wave-uniform branch).
// ---------------------------------------------------------------------------

typedef _Float16 h16  __attribute__((ext_vector_type(16)));
typedef _Float16 h8v  __attribute__((ext_vector_type(8)));
typedef _Float16 h4v  __attribute__((ext_vector_type(4)));
typedef float    f8v  __attribute__((ext_vector_type(8)));
typedef float    f4v  __attribute__((ext_vector_type(4)));

#define SEQ   2048
#define DH    128
#define BQ    128          // queries per workgroup (8 waves * 16)
#define BK    64           // keys per LDS tile (4 x 16-key WMMA tiles)
#define KPAD  136          // padded row (halfs) for K tile: 128 + 8
#define VPAD  72           // padded row (halfs) for V^T tile: 64 + 8
#define NEGINF (-1e30f)

// 1/sqrt(128) * log2(e): fold softmax scale + exp2 conversion into Q.
#define QSCALE (0.08838834764831845f * 1.4426950408889634f)

__global__ __launch_bounds__(256) void fa_fwd_kernel(
    const float* __restrict__ q,
    const float* __restrict__ k,
    const float* __restrict__ v,
    float* __restrict__ out)
{
    __shared__ __align__(16) _Float16 Klds[BK * KPAD];      // [key][d]   f16
    __shared__ __align__(16) _Float16 Vlds[DH * VPAD];      // [d][key]   f16 (transposed)

    const int tid    = threadIdx.x;
    const int lane   = tid & 31;
    const int wave   = tid >> 5;
    const int lane16 = lane & 15;
    const bool lo    = lane < 16;
    const int hi8    = lo ? 0 : 8;
    const int o0     = lo ? 0 : 16;     // A-frag low-run offset (halfs)
    const int o1     = lo ? 8 : 24;     // A-frag high-run offset (halfs)

    const int bh   = blockIdx.y;                 // fused batch*head
    const int qblk = blockIdx.x * BQ;            // block's first query
    const int qw   = qblk + wave * 16;           // wave's first query
    const int qrow = qw + lane16;                // this lane's query (column)

    const float* Qb = q   + (size_t)bh * SEQ * DH;
    const float* Kb = k   + (size_t)bh * SEQ * DH;
    const float* Vb = v   + (size_t)bh * SEQ * DH;
    float*       Ob = out + (size_t)bh * SEQ * DH;

    // ---- Load Q^T B-fragments once (4 chunks of K=32 over d) ---------------
    // B frag (32x16, 16-bit): lane<16 -> column n=lane, K(=d) 0..15 of chunk;
    // lane>=16 -> same column, K 16..31.  That's 16 contiguous f32 per lane.
    h16 qf[4];
    {
        const float* qp = Qb + (size_t)qrow * DH + (lo ? 0 : 16);
#pragma unroll
        for (int c = 0; c < 4; ++c) {
            const float* p = qp + c * 32;
            h16 t;
#pragma unroll
            for (int j = 0; j < 16; ++j)
                t[j] = (_Float16)(p[j] * QSCALE);
            qf[c] = t;
        }
    }

    // ---- Running flash-softmax state --------------------------------------
    f8v acc[8];                 // O^T accum: 8 d-tiles x (8 VGPR f32)
#pragma unroll
    for (int dt = 0; dt < 8; ++dt)
#pragma unroll
        for (int r = 0; r < 8; ++r)
            acc[dt][r] = 0.0f;
    float m    = NEGINF;        // running max (log2 domain) for this lane's query
    float lsum = 0.0f;          // running sum

    const int nkv = (qblk + BQ) / BK;            // causal: only blocks <= diag
    for (int jb = 0; jb < nkv; ++jb) {
        const int kv = jb * BK;

        // ---- Cooperative stage: K tile (row-major) + V tile (transposed) --
#pragma unroll
        for (int i = 0; i < 8; ++i) {
            const int f   = tid + 256 * i;       // 2048 float4's total
            const int row = f >> 5;              // key within tile (0..63)
            const int c4  = (f & 31) * 4;        // d offset
            const f4v kd = *(const f4v*)(Kb + (size_t)(kv + row) * DH + c4);
            const f4v vd = *(const f4v*)(Vb + (size_t)(kv + row) * DH + c4);
            h4v kh;
#pragma unroll
            for (int jj = 0; jj < 4; ++jj) kh[jj] = (_Float16)kd[jj];
            *(h4v*)&Klds[row * KPAD + c4] = kh;
#pragma unroll
            for (int jj = 0; jj < 4; ++jj)
                Vlds[(c4 + jj) * VPAD + row] = (_Float16)vd[jj];
        }
        __syncthreads();

        if (kv <= qw + 15) {     // uniform per wave: any unmasked work here?
            // ---- S^T tiles: ct[t] = keys kv+16t .. kv+16t+15 ---------------
            f8v ct[4];
#pragma unroll
            for (int t = 0; t < 4; ++t) ct[t] = f8v{};
#pragma unroll
            for (int c = 0; c < 4; ++c) {
                // A frag (16x32, 16-bit): lane<16 row m=lane, K 0..15 (same order as qf);
                // lane>=16 same row, K 16..31. Contiguous 8-half runs.
#pragma unroll
                for (int t = 0; t < 4; ++t) {
                    const _Float16* kr = &Klds[(t * 16 + lane16) * KPAD + c * 32];
                    h8v al = *(const h8v*)(kr + o0);
                    h8v ah = *(const h8v*)(kr + o1);
                    h16 a;
#pragma unroll
                    for (int jj = 0; jj < 8; ++jj) { a[jj] = al[jj]; a[8 + jj] = ah[jj]; }
                    ct[t] = __builtin_amdgcn_wmma_f32_16x16x32_f16(
                                false, a, false, qf[c], (short)0, ct[t], false, false);
                }
            }

            // ---- causal mask: only on diagonal blocks (wave-uniform) -------
            if (kv + BK - 1 > qw) {
                const int th = qrow - kv - hi8;      // mask if (t*16 + r) > th
#pragma unroll
                for (int t = 0; t < 4; ++t)
#pragma unroll
                    for (int r = 0; r < 8; ++r)
                        ct[t][r] = ((t * 16 + r) > th) ? NEGINF : ct[t][r];
            }

            // ---- online softmax (log2 domain) ------------------------------
            float rowm = NEGINF;
#pragma unroll
            for (int t = 0; t < 4; ++t)
#pragma unroll
                for (int r = 0; r < 8; ++r)
                    rowm = fmaxf(rowm, ct[t][r]);
            rowm = fmaxf(rowm, __shfl_xor(rowm, 16, 32));
            const float mnew  = fmaxf(m, rowm);
            const float alpha = __builtin_amdgcn_exp2f(m - mnew);
            m = mnew;

            f8v pt[4];
            float psum = 0.0f;
#pragma unroll
            for (int t = 0; t < 4; ++t)
#pragma unroll
                for (int r = 0; r < 8; ++r) {
                    pt[t][r] = __builtin_amdgcn_exp2f(ct[t][r] - mnew);
                    psum += pt[t][r];
                }
            psum += __shfl_xor(psum, 16, 32);
            lsum = lsum * alpha + psum;

            // ---- P^T B-fragments via shfl_xor(16) + select -----------------
            // pb0: keys kv..kv+31 (tiles 0,1); pb1: keys kv+32..kv+63 (2,3).
            h16 pb0, pb1;
#pragma unroll
            for (int r = 0; r < 8; ++r) {
                const float s0 = __shfl_xor(pt[0][r], 16, 32);
                const float s1 = __shfl_xor(pt[1][r], 16, 32);
                const float s2 = __shfl_xor(pt[2][r], 16, 32);
                const float s3 = __shfl_xor(pt[3][r], 16, 32);
                pb0[r]     = (_Float16)(lo ? pt[0][r] : s1);
                pb0[8 + r] = (_Float16)(lo ? s0       : pt[1][r]);
                pb1[r]     = (_Float16)(lo ? pt[2][r] : s3);
                pb1[8 + r] = (_Float16)(lo ? s2       : pt[3][r]);
            }

            // ---- O^T := alpha*O^T + V^T * P^T (two K=32 steps) -------------
#pragma unroll
            for (int dt = 0; dt < 8; ++dt) {
                f8v o = acc[dt];
#pragma unroll
                for (int r = 0; r < 8; ++r) o[r] *= alpha;
                const _Float16* vr = &Vlds[(dt * 16 + lane16) * VPAD];
                h8v v0l = *(const h8v*)(vr + o0);
                h8v v0h = *(const h8v*)(vr + o1);
                h8v v1l = *(const h8v*)(vr + 32 + o0);
                h8v v1h = *(const h8v*)(vr + 32 + o1);
                h16 a0, a1;
#pragma unroll
                for (int jj = 0; jj < 8; ++jj) {
                    a0[jj] = v0l[jj]; a0[8 + jj] = v0h[jj];
                    a1[jj] = v1l[jj]; a1[8 + jj] = v1h[jj];
                }
                o = __builtin_amdgcn_wmma_f32_16x16x32_f16(
                        false, a0, false, pb0, (short)0, o, false, false);
                acc[dt] = __builtin_amdgcn_wmma_f32_16x16x32_f16(
                        false, a1, false, pb1, (short)0, o, false, false);
            }
        }
        __syncthreads();
    }

    // ---- epilogue: O = O^T / lsum, store (lane's query row, d contiguous) --
    const float inv = 1.0f / lsum;
#pragma unroll
    for (int dt = 0; dt < 8; ++dt) {
        float* op = Ob + (size_t)qrow * DH + dt * 16 + hi8;
        f4v s0, s1;
#pragma unroll
        for (int r = 0; r < 4; ++r) { s0[r] = acc[dt][r] * inv; s1[r] = acc[dt][4 + r] * inv; }
        *(f4v*)op       = s0;
        *(f4v*)(op + 4) = s1;
    }
}

extern "C" void kernel_launch(void* const* d_in, const int* in_sizes, int n_in,
                              void* d_out, int out_size, void* d_ws, size_t ws_size,
                              hipStream_t stream) {
    (void)n_in; (void)d_ws; (void)ws_size;
    const float* q = (const float*)d_in[0];
    const float* k = (const float*)d_in[1];
    const float* v = (const float*)d_in[2];
    // d_in[3] is the causal bool mask; causality is computed analytically.
    float* out = (float*)d_out;

    const int BH = in_sizes[0] / (SEQ * DH);     // B*H = 64
    (void)out_size;
    dim3 grid(SEQ / BQ, BH);
    dim3 block(256);
    fa_fwd_kernel<<<grid, block, 0, stream>>>(q, k, v, out);
}
